// GAT_45827301048627
// MI455X (gfx1250) — matrix operand order repaired and checked by reading, not compiled, not run on hardware
//
#include <hip/hip_runtime.h>

// ---------------------------------------------------------------------------
// GAT (2-layer, PyG GATConv semantics) for gfx1250 (MI455X, wave32).
// GEMMs: v_wmma_f32_16x16x32_bf16, LDS double-buffered, A tile staged with
// GLOBAL_LOAD_ASYNC_TO_LDS_B128 (ASYNCcnt) when the builtin exists.
// Edge phase: streaming + f32 atomics (resolve in 192MB L2; h fits in L2).
// ---------------------------------------------------------------------------

typedef __attribute__((ext_vector_type(16))) __bf16 v16bf;
typedef __attribute__((ext_vector_type(8)))  float  v8f;

#define FIN   512
#define HDIM  8
#define CDIM  32
#define HC    256
#define NEG_SLOPE 0.2f

#if defined(__AMDGCN__) && \
    __has_builtin(__builtin_amdgcn_global_load_async_to_lds_b128) && \
    __has_builtin(__builtin_amdgcn_s_wait_asynccnt)
#define USE_ASYNC_COPY 1
// Builtin signature (from hipcc diagnostic): (v4i AS1*, v4i AS3*, Imm, Imm)
typedef int async_v4i __attribute__((vector_size(16)));
typedef __attribute__((address_space(1))) async_v4i* async_gptr;
typedef __attribute__((address_space(3))) async_v4i* async_lptr;
#else
#define USE_ASYNC_COPY 0
#endif

// ------------------------------ helpers -----------------------------------

__device__ __forceinline__ unsigned enc_f32(float x) {
  unsigned u = __float_as_uint(x);
  return (u & 0x80000000u) ? ~u : (u | 0x80000000u);   // order-preserving
}
__device__ __forceinline__ float dec_f32(unsigned e) {
  unsigned u = (e & 0x80000000u) ? (e & 0x7FFFFFFFu) : ~e;
  return __uint_as_float(u);
}

// ------------------------------ cast kernels -------------------------------

__global__ void k_cast_bf16(const float* __restrict__ in,
                            __bf16* __restrict__ out, int n) {
  int i = (blockIdx.x * blockDim.x + threadIdx.x) * 4;
  if (i + 3 < n) {
    float4 v = *(const float4*)(in + i);
    out[i + 0] = (__bf16)v.x;
    out[i + 1] = (__bf16)v.y;
    out[i + 2] = (__bf16)v.z;
    out[i + 3] = (__bf16)v.w;
  }
}

__global__ void k_bias_relu_cast(const float* __restrict__ in,
                                 const float* __restrict__ bias,
                                 __bf16* __restrict__ out, int n) {
  int i = blockIdx.x * blockDim.x + threadIdx.x;
  if (i < n) {
    float v = in[i] + bias[i & (HC - 1)];
    v = v > 0.f ? v : 0.f;
    out[i] = (__bf16)v;
  }
}

__global__ void k_fill_u32(unsigned* __restrict__ p, unsigned v, int n) {
  int i = blockIdx.x * blockDim.x + threadIdx.x;
  if (i < n) p[i] = v;
}

// ------------------------------ WMMA GEMM ----------------------------------
// C[MxNn] (f32) = A[MxK] (bf16, row-major) * B[KxNn] (bf16, row-major).
// Block tile 128x64, 8 waves; each wave: 16 rows x 64 cols (4 WMMA tiles).
// K-step 32, double-buffered LDS. B is stored TRANSPOSED in LDS (Bs[col][k])
// so both A and B fragments are contiguous ds_load_b128 pairs.

__global__ __launch_bounds__(256)
void k_gemm_bf16(const __bf16* __restrict__ A, const __bf16* __restrict__ B,
                 float* __restrict__ C, int M, int K, int Nn) {
  __shared__ __bf16 As[2][128][32];      // [buf][row][k]
  __shared__ __bf16 Bs[2][64][40];       // [buf][col][k], 80B pitch (16B align)

  const int tid  = threadIdx.x;
  const int lane = tid & 31;
  const int wave = tid >> 5;
  const int bn   = blockIdx.x * 64;
  const int bm   = blockIdx.y * 128;
  const int rm   = wave * 16;

  const int ar = tid >> 1;               // A stage: 2 threads/row, 16 bf16 each
  const int ac = (tid & 1) * 16;
  const int br = tid >> 3;               // B stage: 8 threads/row, 8 bf16 each
  const int bc = (tid & 7) * 8;

  // Clamp OOB rows to M-1: a garbage A row r only feeds C row r, never stored.
  int arow = bm + ar;
  if (arow >= M) arow = M - 1;
  const __bf16* agp_base = A + (size_t)arow * K + ac;
  const __bf16* bgp_base = B + (size_t)br * Nn + bn + bc;

  v8f acc[4] = {};

  auto stage = [&](int buf, int kk) {
    // ---- A tile 128x32: async global->LDS (b128 per lane) when available
    const __bf16* agp = agp_base + kk;
    __bf16* alp = &As[buf][ar][ac];
#if USE_ASYNC_COPY
    __builtin_amdgcn_global_load_async_to_lds_b128(
        (async_gptr)agp, (async_lptr)alp, 0, 0);
    __builtin_amdgcn_global_load_async_to_lds_b128(
        (async_gptr)(agp + 8), (async_lptr)(alp + 8), 0, 0);
#else
    const uint4* s = (const uint4*)agp;
    *(uint4*)alp       = s[0];
    *(uint4*)(alp + 8) = s[1];
#endif
    // ---- B tile 32x64: coalesced b128 global load, transpose-scatter to LDS
    union { uint4 u; __bf16 b[8]; } v;
    v.u = *(const uint4*)(bgp_base + (size_t)kk * Nn);
    #pragma unroll
    for (int i = 0; i < 8; ++i) Bs[buf][bc + i][br] = v.b[i];
  };

  stage(0, 0);
#if USE_ASYNC_COPY
  __builtin_amdgcn_s_wait_asynccnt(0);
#endif
  __syncthreads();

  int buf = 0;
  for (int kk = 0; kk < K; kk += 32) {
    if (kk + 32 < K) stage(buf ^ 1, kk + 32);   // prefetch next tile

    // A fragment (16x32 bf16, ISA layout):
    // lanes 0-15: M=lane,    K = {0..7, 16..23}
    // lanes16-31: M=lane-16, K = {8..15, 24..31}
    union { v16bf v; uint4 u[2]; } au;
    {
      int r  = rm + (lane & 15);
      int k0 = (lane >> 4) * 8;
      au.u[0] = *(const uint4*)&As[buf][r][k0];
      au.u[1] = *(const uint4*)&As[buf][r][16 + k0];
    }

    #pragma unroll
    for (int t = 0; t < 4; ++t) {
      // B fragment (32x16): lanes 0-15 K={0..7,16..23}, lanes 16-31 K={8..15,24..31},
      // column = lane&15. Contiguous in transposed Bs.
      union { v16bf v; uint4 u[2]; } bu;
      int c  = t * 16 + (lane & 15);
      int k0 = (lane >> 4) * 8;
      const __bf16* bp = &Bs[buf][c][k0];
      bu.u[0] = *(const uint4*)bp;
      bu.u[1] = *(const uint4*)(bp + 16);

      acc[t] = __builtin_amdgcn_wmma_f32_16x16x32_bf16(
          /*neg_a=*/false, au.v, /*neg_b=*/false, bu.v,
          /*c_mod=*/(short)0, acc[t], /*reuse_a=*/false, /*reuse_b=*/false);
    }

#if USE_ASYNC_COPY
    __builtin_amdgcn_s_wait_asynccnt(0);
#endif
    __syncthreads();
    buf ^= 1;
  }

  // ---- store D: VGPR r -> (lanes0-15: M=r, N=lane) (lanes16-31: M=8+r)
  #pragma unroll
  for (int t = 0; t < 4; ++t) {
    int col   = bn + t * 16 + (lane & 15);
    int rbase = bm + rm + (lane >> 4) * 8;
    #pragma unroll
    for (int r = 0; r < 8; ++r) {
      int row = rbase + r;
      if (row < M) C[(size_t)row * Nn + col] = acc[t][r];
    }
  }
}

// --------------------------- attention coefficients ------------------------
// t = n*8 + h ; h[n, h*32 + c] is exactly h_flat[t*32 + c].

__global__ void k_attn_coeff(const float* __restrict__ h,
                             const float* __restrict__ a_src,
                             const float* __restrict__ a_dst,
                             float* __restrict__ als, float* __restrict__ ald,
                             int NH) {
  int t = blockIdx.x * blockDim.x + threadIdx.x;
  if (t >= NH) return;
  int hh = t & (HDIM - 1);
  const float* hp = h + (size_t)t * CDIM;
  float s = 0.f, d = 0.f;
  #pragma unroll
  for (int c = 0; c < CDIM; c += 4) {
    float4 hv = *(const float4*)(hp + c);
    float4 av = *(const float4*)(a_src + hh * CDIM + c);
    float4 dv = *(const float4*)(a_dst + hh * CDIM + c);
    s += hv.x * av.x + hv.y * av.y + hv.z * av.z + hv.w * av.w;
    d += hv.x * dv.x + hv.y * dv.y + hv.z * dv.z + hv.w * dv.w;
  }
  als[t] = s;
  ald[t] = d;
}

// ------------------------------ edge passes --------------------------------
// Self-loops: edges E..E+N-1 are (i,i).

__global__ void k_edge_max(const long long* __restrict__ ei, int E, int N,
                           const float* __restrict__ als,
                           const float* __restrict__ ald,
                           unsigned* __restrict__ menc) {
  int e = blockIdx.x * blockDim.x + threadIdx.x;
  if (e >= E + N) return;
  long long s, d;
  if (e < E) { s = ei[e]; d = ei[E + e]; } else { s = d = e - E; }
  #pragma unroll
  for (int h = 0; h < HDIM; ++h) {
    float v = als[s * HDIM + h] + ald[d * HDIM + h];
    v = v > 0.f ? v : NEG_SLOPE * v;
    atomicMax(&menc[d * HDIM + h], enc_f32(v));
  }
}

__global__ void k_edge_expsum(const long long* __restrict__ ei, int E, int N,
                              const float* __restrict__ als,
                              const float* __restrict__ ald,
                              const unsigned* __restrict__ menc,
                              float* __restrict__ ew, float* __restrict__ den) {
  int e = blockIdx.x * blockDim.x + threadIdx.x;
  if (e >= E + N) return;
  long long s, d;
  if (e < E) { s = ei[e]; d = ei[E + e]; } else { s = d = e - E; }
  #pragma unroll
  for (int h = 0; h < HDIM; ++h) {
    float v = als[s * HDIM + h] + ald[d * HDIM + h];
    v = v > 0.f ? v : NEG_SLOPE * v;
    float m = dec_f32(menc[d * HDIM + h]);
    float w = expf(v - m);
    __builtin_nontemporal_store(w, &ew[(size_t)e * HDIM + h]);  // stream once
    atomicAdd(&den[d * HDIM + h], w);
  }
}

// 64 threads per edge; each thread handles one (head, c4) float4 chunk.
__global__ void k_aggregate(const long long* __restrict__ ei, int E, int N,
                            const float* __restrict__ hmat,
                            const float* __restrict__ ew,
                            const float* __restrict__ den,
                            float* __restrict__ out) {
  long long t = (long long)blockIdx.x * blockDim.x + threadIdx.x;
  long long e = t >> 6;
  int f = (int)(t & 63);
  if (e >= (long long)E + N) return;
  long long s, d;
  if (e < E) { s = ei[e]; d = ei[E + e]; } else { s = d = e - E; }
  int h  = f >> 3;
  int c4 = (f & 7) * 4;
  float w     = __builtin_nontemporal_load(&ew[e * HDIM + h]);
  float alpha = w / (den[d * HDIM + h] + 1e-16f);
  float4 hv = *(const float4*)(hmat + (size_t)s * HC + h * CDIM + c4);
  float* op = out + (size_t)d * HC + h * CDIM + c4;
  atomicAdd(op + 0, hv.x * alpha);
  atomicAdd(op + 1, hv.y * alpha);
  atomicAdd(op + 2, hv.z * alpha);
  atomicAdd(op + 3, hv.w * alpha);
}

// ------------------------------ epilogue -----------------------------------

__global__ void k_mean_bias(const float* __restrict__ agg,
                            const float* __restrict__ b2,
                            float* __restrict__ out, int N) {
  int t = blockIdx.x * blockDim.x + threadIdx.x;   // n*32 + c
  if (t >= N * CDIM) return;
  int n = t >> 5, c = t & 31;
  float s = 0.f;
  #pragma unroll
  for (int h = 0; h < HDIM; ++h) s += agg[(size_t)n * HC + h * CDIM + c];
  out[t] = s * (1.f / HDIM) + b2[c];
}

__global__ void k_final(const float* __restrict__ x2,
                        const float* __restrict__ Wp,
                        const float* __restrict__ bp,
                        float* __restrict__ out, int N) {
  int n = blockIdx.x * blockDim.x + threadIdx.x;
  if (n >= N) return;
  float s = 0.f;
  #pragma unroll
  for (int c = 0; c < CDIM; ++c) s += x2[(size_t)n * CDIM + c] * Wp[c];
  s += bp[0];
  out[n] = 1.f / (1.f + expf(-s));
}

// ------------------------------ launcher -----------------------------------

extern "C" void kernel_launch(void* const* d_in, const int* in_sizes, int n_in,
                              void* d_out, int out_size, void* d_ws,
                              size_t ws_size, hipStream_t stream) {
  const long long* ei = (const long long*)d_in[0];   // [2,E] int64
  const float* x   = (const float*)d_in[1];
  const float* W1  = (const float*)d_in[2];
  const float* as1 = (const float*)d_in[3];
  const float* ad1 = (const float*)d_in[4];
  const float* b1  = (const float*)d_in[5];
  const float* W2  = (const float*)d_in[6];
  const float* as2 = (const float*)d_in[7];
  const float* ad2 = (const float*)d_in[8];
  const float* b2  = (const float*)d_in[9];
  const float* Wp  = (const float*)d_in[10];
  const float* bp  = (const float*)d_in[11];

  const int E  = in_sizes[0] / 2;
  const int N  = in_sizes[1] / FIN;
  const int EP = E + N;

  // ---- workspace carve-out (256B aligned)
  char* ws = (char*)d_ws;
  size_t off = 0;
  auto alloc = [&](size_t bytes) -> char* {
    char* p = ws + off;
    off += (bytes + 255) & ~(size_t)255;
    return p;
  };
  __bf16*   xb   = (__bf16*)  alloc((size_t)N * FIN * 2);   // bf16 x
  __bf16*   w1b  = (__bf16*)  alloc((size_t)FIN * HC * 2);
  __bf16*   w2b  = (__bf16*)  alloc((size_t)HC * HC * 2);
  float*    h    = (float*)   alloc((size_t)N * HC * 4);    // h1, then h2
  float*    als  = (float*)   alloc((size_t)N * HDIM * 4);
  float*    ald  = (float*)   alloc((size_t)N * HDIM * 4);
  unsigned* menc = (unsigned*)alloc((size_t)N * HDIM * 4);
  float*    den  = (float*)   alloc((size_t)N * HDIM * 4);
  float*    ewb  = (float*)   alloc((size_t)EP * HDIM * 4);
  float*    agg1 = (float*)   alloc((size_t)N * HC * 4);
  __bf16*   hrb  = (__bf16*)  alloc((size_t)N * HC * 2);
  float*    o2m  = (float*)   alloc((size_t)N * CDIM * 4);
  float*    agg2 = (float*)xb;  // alias: xb (N*512*2B) dead after GEMM1,
                                // agg2 needs N*256*4B == same size

  const int TB = 256;
  auto blks = [](long long n, int tb) { return (unsigned)((n + tb - 1) / tb); };

  // ---- casts
  k_cast_bf16<<<blks((long long)N * FIN / 4, TB), TB, 0, stream>>>(x, xb, N * FIN);
  k_cast_bf16<<<blks(FIN * HC / 4, TB), TB, 0, stream>>>(W1, w1b, FIN * HC);
  k_cast_bf16<<<blks(HC * HC / 4, TB), TB, 0, stream>>>(W2, w2b, HC * HC);

  dim3 gg(HC / 64, (N + 127) / 128);

  // =================== layer 1 ===================
  k_gemm_bf16<<<gg, 256, 0, stream>>>(xb, w1b, h, N, FIN, HC);
  k_attn_coeff<<<blks((long long)N * HDIM, TB), TB, 0, stream>>>(h, as1, ad1, als, ald, N * HDIM);
  k_fill_u32<<<blks((long long)N * HDIM, TB), TB, 0, stream>>>(menc, 0u, N * HDIM);
  k_fill_u32<<<blks((long long)N * HDIM, TB), TB, 0, stream>>>((unsigned*)den, 0u, N * HDIM);
  k_fill_u32<<<blks((long long)N * HC, TB), TB, 0, stream>>>((unsigned*)agg1, 0u, N * HC);
  k_edge_max<<<blks(EP, TB), TB, 0, stream>>>(ei, E, N, als, ald, menc);
  k_edge_expsum<<<blks(EP, TB), TB, 0, stream>>>(ei, E, N, als, ald, menc, ewb, den);
  k_aggregate<<<blks((long long)EP * 64, TB), TB, 0, stream>>>(ei, E, N, h, ewb, den, agg1);
  k_bias_relu_cast<<<blks((long long)N * HC, TB), TB, 0, stream>>>(agg1, b1, hrb, N * HC);

  // =================== layer 2 ===================
  k_gemm_bf16<<<gg, 256, 0, stream>>>(hrb, w2b, h, N, HC, HC);
  k_attn_coeff<<<blks((long long)N * HDIM, TB), TB, 0, stream>>>(h, as2, ad2, als, ald, N * HDIM);
  k_fill_u32<<<blks((long long)N * HDIM, TB), TB, 0, stream>>>(menc, 0u, N * HDIM);
  k_fill_u32<<<blks((long long)N * HDIM, TB), TB, 0, stream>>>((unsigned*)den, 0u, N * HDIM);
  k_fill_u32<<<blks((long long)N * HC, TB), TB, 0, stream>>>((unsigned*)agg2, 0u, N * HC);
  k_edge_max<<<blks(EP, TB), TB, 0, stream>>>(ei, E, N, als, ald, menc);
  k_edge_expsum<<<blks(EP, TB), TB, 0, stream>>>(ei, E, N, als, ald, menc, ewb, den);
  k_aggregate<<<blks((long long)EP * 64, TB), TB, 0, stream>>>(ei, E, N, h, ewb, den, agg2);

  // ---- head mean + bias, projection + sigmoid
  k_mean_bias<<<blks((long long)N * CDIM, TB), TB, 0, stream>>>(agg2, b2, o2m, N);
  k_final<<<blks(N, TB), TB, 0, stream>>>(o2m, Wp, bp, (float*)d_out, N);
}
